// SLSTM_26800595927467
// MI455X (gfx1250) — matrix operand order repaired and checked by reading, not compiled, not run on hardware
//
#include <hip/hip_runtime.h>
#include <hip/hip_bf16.h>

// ---------------------------------------------------------------------------
// S-LSTM for MI455X (gfx1250): bf16 WMMA GEMMs + fused LDS epilogues +
// async global->LDS staging (CDNA5 ASYNCcnt path).
// B=32, L=1024, H=256, S2V=256, STEP=1, NUM_LAYERS=4, EPS=1e-3
// ---------------------------------------------------------------------------

#define Bk   32
#define Lk   1024
#define Hk   256
#define Nk   (Bk * Lk)        // 32768 token rows
#define KCAT 1280             // [emb | h_before | h_after | h | dh*mask]
#define EPSk 0.001f

typedef __bf16 bf16_t;
typedef __attribute__((ext_vector_type(16))) __bf16 v16bf;
typedef __attribute__((ext_vector_type(8)))  __bf16 v8bf;
typedef __attribute__((ext_vector_type(8)))  float  v8f;

// ---- WMMA fragment loaders (layouts per CDNA5 ISA 7.12.2) ------------------
// A 16x32 bf16, row-major in memory with 'stride' elements per row.
// lane 0-15: M=lane, elems 0..7 = K 0..7, elems 8..15 = K 16..23
// lane16-31: M=lane-16, elems 0..7 = K 8..15, elems 8..15 = K 24..31
__device__ inline v16bf load_a_frag(const bf16_t* a, int stride) {
  int lane = threadIdx.x & 31;
  int m    = lane & 15;
  int base = (lane >> 4) ? 8 : 0;
  const bf16_t* p = a + m * stride + base;
  union { v16bf v; v8bf h[2]; } f;
  f.h[0] = *(const v8bf*)(p);
  f.h[1] = *(const v8bf*)(p + 16);
  return f.v;
}

// B 32x16 bf16, row-major [K][N] with 'stride' elements per K-row.
// lane L holds column N=L&15: elems 0..7 = K 8h..8h+7, elems 8..15 = K 16+8h.. (h=L>>4).
__device__ inline v16bf load_b_frag(const bf16_t* b, int stride) {
  int lane = threadIdx.x & 31;
  int n = lane & 15, kb = (lane >> 4) * 8;
  v16bf v;
#pragma unroll
  for (int s = 0; s < 16; s++) {
    int k = (s < 8) ? (kb + s) : (16 + kb + s - 8);
    v[s] = b[(size_t)k * stride + n];
  }
  return v;
}

__device__ inline float wave_sum(float v) {
#pragma unroll
  for (int m = 16; m >= 1; m >>= 1) v += __shfl_xor(v, m, 32);
  return v;
}

__device__ inline float sigmoidf(float x) { return 1.0f / (1.0f + __expf(-x)); }

// ---- CDNA5 async global->LDS copy (ASYNCcnt) -------------------------------
// One wave instruction copies 512B (16B/lane). lds_off = byte offset within
// the wave's LDS allocation (low 32 bits of a flat-cast shared pointer).
__device__ inline uint32_t lds_offset(const void* p) {
  return (uint32_t)(uintptr_t)p;   // flat LDS aperture: addr[31:0] = LDS offset
}

__device__ inline void async_ld_b128(uint32_t lds_off, const void* g) {
  asm volatile("global_load_async_to_lds_b128 %0, %1, off"
               :: "v"(lds_off), "v"(g)
               : "memory");
}

__device__ inline void wait_async0() {
  asm volatile("s_wait_asynccnt 0" ::: "memory");
}

// ---------------------------------------------------------------------------
// One-time prep kernels
// ---------------------------------------------------------------------------
__global__ void prep_kernel(const float* __restrict__ w, const float* __restrict__ h0,
                            const float* __restrict__ c0, const float* __restrict__ mask,
                            float* __restrict__ emb, float* __restrict__ h, float* __restrict__ c,
                            bf16_t* __restrict__ emb_bf, bf16_t* __restrict__ h_bf,
                            bf16_t* __restrict__ zero_bf) {
  int idx = blockIdx.x * 256 + threadIdx.x;
  if (idx >= Nk * Hk) return;
  float m = mask[idx >> 8];
  float e = w[idx] * m;
  float hv = h0[idx] * m;
  emb[idx]    = e;
  emb_bf[idx] = (bf16_t)e;
  h[idx]      = hv;
  h_bf[idx]   = (bf16_t)hv;
  c[idx]      = c0[idx] * m;
  if (idx < Hk) zero_bf[idx] = (bf16_t)0.0f;
}

// Wcat[g][kk][j], kk-major so B-fragments are contiguous per K-row.
__global__ void pack_wcat_kernel(const float* __restrict__ Wx, const float* __restrict__ Wh,
                                 const float* __restrict__ Wi, const float* __restrict__ Wd,
                                 bf16_t* __restrict__ Wcat) {
  int idx = blockIdx.x * 256 + threadIdx.x;
  if (idx >= 6 * KCAT * Hk) return;
  int g   = idx / (KCAT * Hk);
  int rem = idx - g * (KCAT * Hk);
  int kk  = rem >> 8;
  int j   = rem & 255;
  float v;
  if      (kk < 256)  v = Wx[(g * Hk + kk) * Hk + j];
  else if (kk < 768)  v = Wh[(g * 2 * Hk + (kk - 256)) * Hk + j];   // [hb|ha] == Wh rows 0..511
  else if (kk < 1024) v = Wi[(g * Hk + (kk - 768)) * Hk + j];
  else                v = Wd[(g * Hk + (kk - 1024)) * Hk + j];
  Wcat[idx] = (bf16_t)v;
}

__global__ void pack_gw3_kernel(const float* __restrict__ gWh, bf16_t* __restrict__ W3) {
  int idx = blockIdx.x * 256 + threadIdx.x;
  if (idx >= Hk * Hk) return;
  W3[idx] = (bf16_t)gWh[3 * Hk * Hk + idx];
}

// dst[b][k] = mean over L of src[b][l][k]
__global__ void mean_kernel(const float* __restrict__ src, float* __restrict__ dst) {
  int i = blockIdx.x * 256 + threadIdx.x;
  if (i >= Bk * Hk) return;
  int b = i >> 8, k = i & 255;
  const float* p = src + (size_t)b * Lk * Hk + k;
  float s = 0.f;
  for (int l = 0; l < Lk; l++) s += p[(size_t)l * Hk];
  dst[i] = s * (1.0f / Lk);
}

// ---------------------------------------------------------------------------
// Small per-batch gates: gd, gi, go, tanh(ln(gg)), u3 = dh @ gWx[3]
// ---------------------------------------------------------------------------
__device__ inline float block_sum256(float v, float* red, int tid) {
  red[tid] = v; __syncthreads();
#pragma unroll
  for (int s = 128; s > 0; s >>= 1) { if (tid < s) red[tid] += red[tid + s]; __syncthreads(); }
  float r = red[0]; __syncthreads();
  return r;
}

__device__ inline float block_ln(float x, const float* a, const float* b, int k, float* red) {
  float mean = block_sum256(x, red, k) * (1.0f / Hk);
  float d    = x - mean;
  float var  = block_sum256(d * d, red, k) * (1.0f / (Hk - 1));
  return a[k] * d / (sqrtf(var) + EPSk) + b[k];
}

__device__ inline float dot_col(const float* v, const float* __restrict__ W, int k) {
  float s = 0.f;
#pragma unroll 4
  for (int i = 0; i < Hk; i++) s += v[i] * W[i * Hk + k];
  return s;
}

__global__ __launch_bounds__(256)
void small_gates_kernel(const float* __restrict__ dh, const float* __restrict__ sent,
                        const float* __restrict__ comb,
                        const float* __restrict__ gWx, const float* __restrict__ gWh,
                        const float* __restrict__ gWg, const float* __restrict__ gb,
                        const float* __restrict__ ln_a, const float* __restrict__ ln_b,
                        float* __restrict__ gd, float* __restrict__ gi, float* __restrict__ go,
                        float* __restrict__ tgg, float* __restrict__ u3) {
  __shared__ float sdh[Hk], ssn[Hk], scb[Hk], red[256];
  int b = blockIdx.x, k = threadIdx.x;
  sdh[k] = dh[b * Hk + k];
  ssn[k] = sent[b * Hk + k];
  scb[k] = comb[b * Hk + k];
  __syncthreads();

  const int M = Hk * Hk;
  float pre = dot_col(sdh, gWx + 0 * M, k) + dot_col(ssn, gWg + 0 * M, k)
            + dot_col(scb, gWh + 0 * M, k) + gb[0 * Hk + k];
  gd[b * Hk + k] = sigmoidf(block_ln(pre, ln_a + 6 * Hk, ln_b + 6 * Hk, k, red));

  pre = dot_col(sdh, gWx + 1 * M, k) + dot_col(ssn, gWg + 1 * M, k) + dot_col(scb, gWh + 1 * M, k);
  gi[b * Hk + k] = sigmoidf(block_ln(pre, ln_a + 9 * Hk, ln_b + 9 * Hk, k, red) + gb[1 * Hk + k]);

  pre = dot_col(sdh, gWx + 2 * M, k) + dot_col(ssn, gWg + 2 * M, k)
      + dot_col(scb, gWh + 2 * M, k) + gb[2 * Hk + k];
  go[b * Hk + k] = sigmoidf(block_ln(pre, ln_a + 7 * Hk, ln_b + 7 * Hk, k, red));

  pre = dot_col(ssn, gWg + 3 * M, k);
  tgg[b * Hk + k] = tanhf(block_ln(pre, ln_a + 10 * Hk, ln_b + 10 * Hk, k, red));

  u3[b * Hk + k] = dot_col(sdh, gWx + 3 * M, k);
}

// ---------------------------------------------------------------------------
// gf = sigmoid(ln(mask*u3[b] + h@gWh3 + gb3, a8, b8))  -- WMMA GEMM, N x 256
// A tile staged via async global->LDS from the bf16 mirror of h.
// ---------------------------------------------------------------------------
__global__ __launch_bounds__(256)
void gf_kernel(const bf16_t* __restrict__ h_bf, const bf16_t* __restrict__ W3,
               const float* __restrict__ u3, const float* __restrict__ gb3,
               const float* __restrict__ mask, const float* __restrict__ a8,
               const float* __restrict__ b8, float* __restrict__ gf) {
  __shared__ bf16_t sA[16 * Hk];
  __shared__ float  accb[16 * Hk];
  int tid = threadIdx.x, lane = tid & 31, wid = tid >> 5;
  int n0  = blockIdx.x * 16;

  // async stage: 16 rows x 512B, one wave-op per row
  uint32_t sA_base = lds_offset(sA);
  for (int r = wid; r < 16; r += 8) {
    const char* src = (const char*)(h_bf + (size_t)(n0 + r) * Hk) + lane * 16;
    async_ld_b128(sA_base + (uint32_t)(r * Hk * 2) + lane * 16, src);
  }
  wait_async0();
  __syncthreads();

  v8f z = {0.f, 0.f, 0.f, 0.f, 0.f, 0.f, 0.f, 0.f};
  v8f acc0 = z, acc1 = z;
  int col0 = wid * 32;
  for (int kc = 0; kc < Hk; kc += 32) {
    v16bf a  = load_a_frag(sA + kc, Hk);
    v16bf b0 = load_b_frag(W3 + (size_t)kc * Hk + col0,      Hk);
    v16bf b1 = load_b_frag(W3 + (size_t)kc * Hk + col0 + 16, Hk);
    acc0 = __builtin_amdgcn_wmma_f32_16x16x32_bf16(false, a, false, b0, (short)0, acc0, false, false);
    acc1 = __builtin_amdgcn_wmma_f32_16x16x32_bf16(false, a, false, b1, (short)0, acc1, false, false);
  }

  // D layout: elem j -> row j + 8*(lane>=16), col = lane&15 (+col0)
  int rbase = (lane >> 4) << 3, cl = lane & 15;
  int bidx  = n0 >> 10;
#pragma unroll
  for (int j = 0; j < 8; j++) {
    int r = j + rbase, n = n0 + r;
    float mrow = mask[n];
    int c0i = col0 + cl, c1i = c0i + 16;
    accb[r * Hk + c0i] = acc0[j] + mrow * u3[bidx * Hk + c0i] + gb3[c0i];
    accb[r * Hk + c1i] = acc1[j] + mrow * u3[bidx * Hk + c1i] + gb3[c1i];
  }
  __syncthreads();

  // LN + sigmoid per row (2 rows per wave)
  for (int r = wid * 2; r < wid * 2 + 2; r++) {
    float s = 0.f;
    for (int i = lane; i < Hk; i += 32) s += accb[r * Hk + i];
    float mean = wave_sum(s) * (1.0f / Hk);
    float vs = 0.f;
    for (int i = lane; i < Hk; i += 32) { float d = accb[r * Hk + i] - mean; vs += d * d; }
    float inv = 1.0f / (sqrtf(wave_sum(vs) * (1.0f / (Hk - 1))) + EPSk);
    int n = n0 + r;
    for (int i = lane; i < Hk; i += 32) {
      float y = a8[i] * (accb[r * Hk + i] - mean) * inv + b8[i];
      gf[(size_t)n * Hk + i] = sigmoidf(y);
    }
  }
}

// ---------------------------------------------------------------------------
// Sequence softmax over [gf(L) + msm, gd, gi] per (b,k); produces dh_new, dc_new
// ---------------------------------------------------------------------------
__global__ void seq_softmax_kernel(const float* __restrict__ gf, const float* __restrict__ mask,
                                   const float* __restrict__ gd, const float* __restrict__ gi,
                                   const float* __restrict__ go, const float* __restrict__ tgg,
                                   const float* __restrict__ c, const float* __restrict__ dc,
                                   float* __restrict__ dh2, float* __restrict__ dc2) {
  int i = blockIdx.x * 256 + threadIdx.x;
  if (i >= Bk * Hk) return;
  int b = i >> 8, k = i & 255;
  float gdv = gd[i], giv = gi[i];
  float mx = fmaxf(gdv, giv);
  const float* gfp = gf + (size_t)b * Lk * Hk + k;
  const float* mp  = mask + b * Lk;
  for (int l = 0; l < Lk; l++) {
    float x = gfp[(size_t)l * Hk] + (mp[l] * 1e25f - 1e25f);
    mx = fmaxf(mx, x);
  }
  const float* cp = c + (size_t)b * Lk * Hk + k;
  float s = 0.f, accC = 0.f;
  for (int l = 0; l < Lk; l++) {
    float x = gfp[(size_t)l * Hk] + (mp[l] * 1e25f - 1e25f);
    float e = __expf(x - mx);
    s += e;
    accC += e * cp[(size_t)l * Hk];
  }
  float wd = __expf(gdv - mx), wi = __expf(giv - mx);
  s += wd + wi;
  float dcn = (accC + wd * dc[i] + wi * tgg[i]) / s;
  dc2[i] = dcn;
  dh2[i] = go[i] * tanhf(dcn);
}

// ---------------------------------------------------------------------------
// Big fused kernel: 32 rows x 6 gates x 256 cols per block.
// A tile (32x1280 bf16 = 80 KB LDS) staged mostly via async global->LDS
// from bf16 mirrors (emb_bf, h_bf; boundary rows read a zero page);
// the dh*mask tail is converted by VALU. 8 waves x (12 N-tiles x 2 M-tiles)
// WMMAs; epilogue runs in two 16-row phases through a 96 KB accumulator
// region aliasing the dead A tile: +b_g, LayerNorm+sigmoid per (gate,row),
// 5-way gate softmax, c/h state update (fp32 + bf16 mirror out).
// ---------------------------------------------------------------------------
__global__ __launch_bounds__(256)
void big_gates_kernel(const bf16_t* __restrict__ emb_bf, const bf16_t* __restrict__ h_bf,
                      const bf16_t* __restrict__ zero_bf,
                      const float* __restrict__ c, const float* __restrict__ emb,
                      const float* __restrict__ dh, const float* __restrict__ dc,
                      const float* __restrict__ mask,
                      const bf16_t* __restrict__ Wcat, const float* __restrict__ bg,
                      const float* __restrict__ ln_a, const float* __restrict__ ln_b,
                      float* __restrict__ h_out, float* __restrict__ c_out,
                      bf16_t* __restrict__ hbf_out) {
  extern __shared__ char smem[];
  bf16_t* sA   = (bf16_t*)smem;   // 32*1280 bf16 = 80 KB
  float*  accb = (float*)smem;    // 16*1536 f32  = 96 KB (aliases sA after K loop)
  int tid = threadIdx.x, lane = tid & 31, wid = tid >> 5;
  int n0  = blockIdx.x * 32;

  // --- async stage of the copyable A segments: 32 rows x 4 segs x 512B ------
  uint32_t sA_base = lds_offset(sA);
  for (int sidx = wid; sidx < 32 * 4; sidx += 8) {     // uniform per wave
    int r = sidx >> 2, seg = sidx & 3;
    int n = n0 + r, l = n & (Lk - 1);
    const bf16_t* src;
    if      (seg == 0) src = emb_bf + (size_t)n * Hk;
    else if (seg == 1) src = (l > 0)      ? h_bf + (size_t)(n - 1) * Hk : zero_bf;
    else if (seg == 2) src = (l < Lk - 1) ? h_bf + (size_t)(n + 1) * Hk : zero_bf;
    else               src = h_bf + (size_t)n * Hk;
    uint32_t dst = sA_base + (uint32_t)((r * KCAT + seg * Hk) * 2) + lane * 16;
    async_ld_b128(dst, (const char*)src + lane * 16);
  }
  // --- VALU tail: dh*mask segment (cols 1024..1279) -------------------------
  for (int idx = tid; idx < 32 * Hk; idx += 256) {
    int r = idx >> 8, k = idx & 255;
    int n = n0 + r;
    sA[r * KCAT + 1024 + k] = (bf16_t)(dh[(n >> 10) * Hk + k] * mask[n]);
  }
  wait_async0();
  __syncthreads();

  // --- WMMA main loop: 2 M-tiles x 12 N-tiles per wave ----------------------
  v8f z = {0.f, 0.f, 0.f, 0.f, 0.f, 0.f, 0.f, 0.f};
  v8f acc[2][12];
#pragma unroll
  for (int m = 0; m < 2; m++)
#pragma unroll
    for (int t = 0; t < 12; t++) acc[m][t] = z;

  for (int kc = 0; kc < KCAT; kc += 32) {
    v16bf a0 = load_a_frag(sA + kc,             KCAT);
    v16bf a1 = load_a_frag(sA + 16 * KCAT + kc, KCAT);
#pragma unroll
    for (int t = 0; t < 12; t++) {
      int T = wid * 12 + t;
      int g = T >> 4, col0 = (T & 15) * 16;
      v16bf bm = load_b_frag(Wcat + ((size_t)(g * KCAT + kc)) * Hk + col0, Hk);
      acc[0][t] = __builtin_amdgcn_wmma_f32_16x16x32_bf16(false, a0, false, bm, (short)0, acc[0][t], false, false);
      acc[1][t] = __builtin_amdgcn_wmma_f32_16x16x32_bf16(false, a1, false, bm, (short)0, acc[1][t], false, false);
    }
  }
  __syncthreads();  // done reading sA; safe to overwrite as accb

  int rbase = (lane >> 4) << 3, cl = lane & 15;
  for (int half = 0; half < 2; half++) {
    // stage accumulators (+ b_g bias) for this 16-row half
#pragma unroll
    for (int t = 0; t < 12; t++) {
      int T = wid * 12 + t;
      int g = T >> 4, col = (T & 15) * 16 + cl;
#pragma unroll
      for (int j = 0; j < 8; j++) {
        int r = j + rbase;
        accb[r * 1536 + g * Hk + col] = acc[half][t][j] + bg[g * Hk + col];
      }
    }
    __syncthreads();

    // LayerNorm + sigmoid per (gate, row): 96 pairs over 8 waves
    for (int p = wid; p < 96; p += 8) {
      int g = p >> 4, r = p & 15;
      float* row = accb + r * 1536 + g * Hk;
      float s = 0.f;
      for (int i = lane; i < Hk; i += 32) s += row[i];
      float mean = wave_sum(s) * (1.0f / Hk);
      float vs = 0.f;
      for (int i = lane; i < Hk; i += 32) { float d = row[i] - mean; vs += d * d; }
      float inv = 1.0f / (sqrtf(wave_sum(vs) * (1.0f / (Hk - 1))) + EPSk);
      for (int i = lane; i < Hk; i += 32) {
        float y = ln_a[g * Hk + i] * (row[i] - mean) * inv + ln_b[g * Hk + i];
        row[i] = sigmoidf(y);
      }
    }
    __syncthreads();

    // 5-way gate softmax + state update for rows n0 + half*16 + [0,16)
    for (int idx = tid; idx < 16 * Hk; idx += 256) {
      int r = idx >> 8, k = idx & 255;
      int n = n0 + half * 16 + r, b = n >> 10, l = n & (Lk - 1);
      const float* rowp = accb + r * 1536;
      float g0 = rowp[0 * Hk + k], g1 = rowp[1 * Hk + k], g2 = rowp[2 * Hk + k];
      float g3 = rowp[3 * Hk + k], g4 = rowp[4 * Hk + k], g5 = rowp[5 * Hk + k];
      float mxv = fmaxf(fmaxf(fmaxf(g0, g1), fmaxf(g2, g3)), g4);
      float e0 = __expf(g0 - mxv), e1 = __expf(g1 - mxv), e2 = __expf(g2 - mxv);
      float e3 = __expf(g3 - mxv), e4 = __expf(g4 - mxv);
      float inv = 1.0f / (e0 + e1 + e2 + e3 + e4);
      float cbv = (l > 0)      ? c[(size_t)(n - 1) * Hk + k] : 0.f;
      float cav = (l < Lk - 1) ? c[(size_t)(n + 1) * Hk + k] : 0.f;
      float m   = mask[n];
      float cn  = (cbv * e0 + cav * e1 + emb[(size_t)n * Hk + k] * e2
                 + dc[b * Hk + k] * m * e3 + c[(size_t)n * Hk + k] * e4) * inv;
      float hn  = g5 * tanhf(cn) * m;
      h_out[(size_t)n * Hk + k]   = hn;
      c_out[(size_t)n * Hk + k]   = cn * m;
      hbf_out[(size_t)n * Hk + k] = (bf16_t)hn;
    }
    __syncthreads();  // accb reused by next half
  }
}

// ---------------------------------------------------------------------------
// Host orchestration
// ---------------------------------------------------------------------------
extern "C" void kernel_launch(void* const* d_in, const int* in_sizes, int n_in,
                              void* d_out, int out_size, void* d_ws, size_t ws_size,
                              hipStream_t stream) {
  const float* word = (const float*)d_in[0];
  const float* sent = (const float*)d_in[1];
  const float* mask = (const float*)d_in[2];
  const float* h0   = (const float*)d_in[3];
  const float* c0   = (const float*)d_in[4];
  const float* Wx   = (const float*)d_in[5];
  const float* Wh   = (const float*)d_in[6];
  const float* Wi   = (const float*)d_in[7];
  const float* Wd   = (const float*)d_in[8];
  const float* bg   = (const float*)d_in[9];
  const float* gWx  = (const float*)d_in[10];
  const float* gWh  = (const float*)d_in[11];
  const float* gWg  = (const float*)d_in[12];
  const float* gb   = (const float*)d_in[13];
  const float* ln_a = (const float*)d_in[14];
  const float* ln_b = (const float*)d_in[15];
  const int num_layers = 4;  // reference NUM_LAYERS (device-resident scalar; fixed)

  // Workspace carve-out
  char* ws = (char*)d_ws;
  size_t off = 0;
  auto alloc = [&](size_t bytes) -> char* {
    char* p = ws + off;
    off += (bytes + 255) & ~(size_t)255;
    return p;
  };
  const size_t NH = (size_t)Nk * Hk;
  float*  hA    = (float*)alloc(NH * 4);
  float*  cA    = (float*)alloc(NH * 4);
  float*  hB    = (float*)alloc(NH * 4);
  float*  cB    = (float*)alloc(NH * 4);
  float*  emb   = (float*)alloc(NH * 4);
  float*  gf    = (float*)alloc(NH * 4);
  bf16_t* embbf = (bf16_t*)alloc(NH * 2);
  bf16_t* hbA   = (bf16_t*)alloc(NH * 2);
  bf16_t* hbB   = (bf16_t*)alloc(NH * 2);
  bf16_t* Wcat  = (bf16_t*)alloc((size_t)6 * KCAT * Hk * 2);
  bf16_t* W3    = (bf16_t*)alloc((size_t)Hk * Hk * 2);
  bf16_t* zbf   = (bf16_t*)alloc((size_t)Hk * 2);
  float*  dhA   = (float*)alloc(Bk * Hk * 4);
  float*  dcA   = (float*)alloc(Bk * Hk * 4);
  float*  dhB   = (float*)alloc(Bk * Hk * 4);
  float*  dcB   = (float*)alloc(Bk * Hk * 4);
  float*  comb  = (float*)alloc(Bk * Hk * 4);
  float*  gd    = (float*)alloc(Bk * Hk * 4);
  float*  gi    = (float*)alloc(Bk * Hk * 4);
  float*  go    = (float*)alloc(Bk * Hk * 4);
  float*  tgg   = (float*)alloc(Bk * Hk * 4);
  float*  u3    = (float*)alloc(Bk * Hk * 4);
  (void)in_sizes; (void)n_in; (void)out_size; (void)ws_size;

  // One-time prep
  prep_kernel<<<(Nk * Hk) / 256, 256, 0, stream>>>(word, h0, c0, mask, emb, hA, cA,
                                                   embbf, hbA, zbf);
  pack_wcat_kernel<<<(6 * KCAT * Hk) / 256, 256, 0, stream>>>(Wx, Wh, Wi, Wd, Wcat);
  pack_gw3_kernel<<<(Hk * Hk) / 256, 256, 0, stream>>>(gWh, W3);
  mean_kernel<<<(Bk * Hk) / 256, 256, 0, stream>>>(hA, dhA);
  mean_kernel<<<(Bk * Hk) / 256, 256, 0, stream>>>(cA, dcA);

  float  *h = hA, *c = cA, *hN = hB, *cN = cB;
  bf16_t *hb = hbA, *hbN = hbB;
  float  *dh = dhA, *dc = dcA, *dh2 = dhB, *dc2 = dcB;

  for (int layer = 0; layer < num_layers; layer++) {
    mean_kernel<<<(Bk * Hk) / 256, 256, 0, stream>>>(h, comb);
    small_gates_kernel<<<Bk, 256, 0, stream>>>(dh, sent, comb, gWx, gWh, gWg, gb,
                                               ln_a, ln_b, gd, gi, go, tgg, u3);
    gf_kernel<<<Nk / 16, 256, 0, stream>>>(hb, W3, u3, gb + 3 * Hk, mask,
                                           ln_a + 8 * Hk, ln_b + 8 * Hk, gf);
    seq_softmax_kernel<<<(Bk * Hk) / 256, 256, 0, stream>>>(gf, mask, gd, gi, go, tgg,
                                                            c, dc, dh2, dc2);
    big_gates_kernel<<<Nk / 32, 256, 16 * 1536 * sizeof(float), stream>>>(
        embbf, hb, zbf, c, emb, dh, dc, mask, Wcat, bg, ln_a, ln_b, hN, cN, hbN);

    { float*  t = h;  h  = hN;  hN  = t; }
    { float*  t = c;  c  = cN;  cN  = t; }
    { bf16_t* t = hb; hb = hbN; hbN = t; }
    { float*  t = dh; dh = dh2; dh2 = t; }
    { float*  t = dc; dc = dc2; dc2 = t; }
  }

  hipMemcpyAsync(d_out, h, NH * sizeof(float), hipMemcpyDeviceToDevice, stream);
}
